// DynamicSparseLinearAttention_59717225284214
// MI455X (gfx1250) — matrix operand order repaired and checked by reading, not compiled, not run on hardware
//
#include <hip/hip_runtime.h>
#include <hip/hip_bf16.h>

// DynamicSparseLinearAttention for MI455X (gfx1250), wave32 + WMMA bf16.
// out[n,l,h,v] = Z[n,l,h] * sum_d Qfm[n,l,h,d] * KV[n,h,d,v]
//   KV    = sum_s Kfm[s,d] * vals[s,v]        (1/L and *L cancel)
//   score = Qfm . Ksum ;  Z = 1/( (score>thr ? score : 0) + 1e-6 )

typedef __attribute__((ext_vector_type(16))) __bf16 bf16x16;
typedef __attribute__((ext_vector_type(2)))  __bf16 bf16x2;
typedef __attribute__((ext_vector_type(8)))  float  f32x8;
typedef int v4i __attribute__((vector_size(16)));   // async-LDS builtin pointee

#define N_B   4
#define L_S   8192
#define H_H   8
#define D_D   128
#define NH    (N_B * H_H)
#define SPLIT 8
#define CHUNK (L_S / SPLIT)      // 1024 s-values per pass-1 block
#define KSTR  40                 // LDS row stride (halves) for 32-wide s tiles
#define KVSTR 136                // LDS row stride (halves) for 128-wide d rows

#if defined(__gfx1250__) && __has_builtin(__builtin_amdgcn_global_load_async_to_lds_b128)
#define USE_ASYNC_LDS 1
#else
#define USE_ASYNC_LDS 0
#endif

__device__ __forceinline__ float fmap(float x) {       // elu(x)+1
  return x > 0.0f ? x + 1.0f : __expf(x);
}
__device__ __forceinline__ unsigned short f2bf(float f) {  // f32 -> bf16 RNE
  unsigned int u = __float_as_uint(f);
  u += 0x7FFFu + ((u >> 16) & 1u);
  return (unsigned short)(u >> 16);
}
// packed f32x2 -> bf16x2 (hardware cvt when available)
__device__ __forceinline__ unsigned int pack_bf16(float lo, float hi) {
#if __has_builtin(__builtin_amdgcn_cvt_pk_bf16_f32)
  union { bf16x2 v; unsigned int u; } c;
  c.v = __builtin_amdgcn_cvt_pk_bf16_f32(lo, hi);
  return c.u;
#else
  return (unsigned int)f2bf(lo) | ((unsigned int)f2bf(hi) << 16);
#endif
}

union Frag { int4 i2[2]; bf16x16 v; };

// ---------------------------------------------------------------------------
// Pass 1: partial KV (128x128) + partial Ksum over an L-chunk, per (split,nh).
// ---------------------------------------------------------------------------
__global__ __launch_bounds__(256)
void kv_partial_kernel(const float* __restrict__ Kg, const float* __restrict__ Vg,
                       float* __restrict__ kv_part, float* __restrict__ ksum_part) {
  __shared__ unsigned short lds_kT[128 * KSTR];   // Kfm^T : [d][s] bf16
  __shared__ unsigned short lds_vT[128 * KSTR];   // vals^T: [v][s] bf16
  __shared__ float4 lds_red[256];

  const int t    = threadIdx.x;
  const int spl  = blockIdx.x;
  const int nh   = blockIdx.y;
  const int n    = nh >> 3, h = nh & 7;
  const int lane = t & 31, wave = t >> 5;
  const int half = lane >> 4, mr = lane & 15;

  f32x8 acc[8];
  #pragma unroll
  for (int i = 0; i < 8; ++i) acc[i] = f32x8{};

  float4 ks = make_float4(0.f, 0.f, 0.f, 0.f);
  const int d4    = (t & 31) * 4;   // staging: this thread's 4 d/v columns
  const int srow0 = t >> 5;         // staging: s row base

  for (int it = 0; it < CHUNK / 32; ++it) {
    const int s0 = spl * CHUNK + it * 32;
    // ---- stage 32x128 tiles of Kfm and vals into LDS, transposed bf16 ----
    #pragma unroll
    for (int i = 0; i < 4; ++i) {
      const int s = srow0 + 8 * i;  // 0..31
      const int goff = ((n * L_S + (s0 + s)) * H_H + h) * D_D + d4;
      float4 kq = *(const float4*)(Kg + goff);
      float4 vq = *(const float4*)(Vg + goff);
      float f0 = fmap(kq.x), f1 = fmap(kq.y), f2 = fmap(kq.z), f3 = fmap(kq.w);
      ks.x += f0; ks.y += f1; ks.z += f2; ks.w += f3;
      const unsigned int k01 = pack_bf16(f0, f1);
      const unsigned int k23 = pack_bf16(f2, f3);
      const unsigned int v01 = pack_bf16(vq.x, vq.y);
      const unsigned int v23 = pack_bf16(vq.z, vq.w);
      lds_kT[(d4 + 0) * KSTR + s] = (unsigned short)(k01);
      lds_kT[(d4 + 1) * KSTR + s] = (unsigned short)(k01 >> 16);
      lds_kT[(d4 + 2) * KSTR + s] = (unsigned short)(k23);
      lds_kT[(d4 + 3) * KSTR + s] = (unsigned short)(k23 >> 16);
      lds_vT[(d4 + 0) * KSTR + s] = (unsigned short)(v01);
      lds_vT[(d4 + 1) * KSTR + s] = (unsigned short)(v01 >> 16);
      lds_vT[(d4 + 2) * KSTR + s] = (unsigned short)(v23);
      lds_vT[(d4 + 3) * KSTR + s] = (unsigned short)(v23 >> 16);
    }
    __syncthreads();
    // ---- WMMA: wave computes its 16(d) x 128(v) strip of KV ----
    Frag a;
    const unsigned short* ar = &lds_kT[(wave * 16 + mr) * KSTR + half * 8];
    a.i2[0] = *(const int4*)(ar);        // k = half*8 .. +7
    a.i2[1] = *(const int4*)(ar + 16);   // k = 16+half*8 .. +7
    #pragma unroll
    for (int nt = 0; nt < 8; ++nt) {
      Frag b;
      const unsigned short* br = &lds_vT[(nt * 16 + mr) * KSTR + half * 8];
      b.i2[0] = *(const int4*)(br);      // k = half*8 .. +7
      b.i2[1] = *(const int4*)(br + 16); // k = 16+half*8 .. +7
      acc[nt] = __builtin_amdgcn_wmma_f32_16x16x32_bf16(
          false, a.v, false, b.v, (short)0, acc[nt], false, false);
    }
    __syncthreads();
  }

  // write partial KV: [split][nh][d][v] f32
  float* out = kv_part + (size_t)(spl * NH + nh) * (128 * 128);
  #pragma unroll
  for (int nt = 0; nt < 8; ++nt) {
    #pragma unroll
    for (int r = 0; r < 8; ++r) {
      const int M  = wave * 16 + half * 8 + r;   // d row
      const int Vc = nt * 16 + mr;               // v col
      out[M * 128 + Vc] = acc[nt][r];
    }
  }
  // Ksum partial: threads with equal (t&31) share the same 4 d's
  lds_red[t] = ks;
  __syncthreads();
  if (t < 32) {
    float4 s = lds_red[t];
    #pragma unroll
    for (int j = 1; j < 8; ++j) {
      float4 o = lds_red[t + 32 * j];
      s.x += o.x; s.y += o.y; s.z += o.z; s.w += o.w;
    }
    float* kp = ksum_part + (spl * NH + nh) * 128 + t * 4;
    kp[0] = s.x; kp[1] = s.y; kp[2] = s.z; kp[3] = s.w;
  }
}

// ---------------------------------------------------------------------------
// Pass 1b: deterministic reduction over SPLIT; emit KV^T in bf16 + Ksum f32.
// ---------------------------------------------------------------------------
__global__ __launch_bounds__(256)
void reduce_kernel(const float* __restrict__ kv_part, const float* __restrict__ ksum_part,
                   unsigned short* __restrict__ kvT, float* __restrict__ ksum) {
  const int nh = blockIdx.x;
  const int t  = threadIdx.x;
  for (int i = t; i < 128 * 128; i += 256) {
    const int d = i >> 7, v = i & 127;
    float s = 0.f;
    #pragma unroll
    for (int sp = 0; sp < SPLIT; ++sp)
      s += kv_part[(size_t)(sp * NH + nh) * (128 * 128) + i];
    kvT[(size_t)nh * (128 * 128) + v * 128 + d] = f2bf(s);
  }
  if (t < 128) {
    float s = 0.f;
    #pragma unroll
    for (int sp = 0; sp < SPLIT; ++sp) s += ksum_part[(sp * NH + nh) * 128 + t];
    ksum[nh * 128 + t] = s;
  }
}

// ---------------------------------------------------------------------------
// Pass 2: out = Z * (Qfm @ KV). Score computed inline during A-fragment build.
// KV^T staging uses CDNA5 async global->LDS b128 transfers when available.
// ---------------------------------------------------------------------------
__global__ __launch_bounds__(256)
void attn_out_kernel(const float* __restrict__ Qg, const unsigned short* __restrict__ kvTg,
                     const float* __restrict__ ksum, const float* __restrict__ thr,
                     float* __restrict__ Og) {
  __shared__ unsigned short lds_kv[128 * KVSTR];  // KV^T: [v][d] bf16
  __shared__ float lds_ks[128];

  const int t    = threadIdx.x;
  const int nh   = blockIdx.y;
  const int n    = nh >> 3, h = nh & 7;
  const int lane = t & 31, wave = t >> 5;
  const int half = lane >> 4, mr = lane & 15;
  const int Lbase = blockIdx.x * 128;

  // ---- stage KV^T (32KB, pure copy) + Ksum ----
  const unsigned short* src = kvTg + (size_t)nh * (128 * 128);
#if USE_ASYNC_LDS
  {
    // Async DMA into LDS: 2048 x 16B chunks. Global row = 16 chunks (256B);
    // padded LDS row = 17 chunks (272B). Tracked by ASYNCcnt, no VGPR hop.
    __attribute__((address_space(1))) v4i* g =
        (__attribute__((address_space(1))) v4i*)(uintptr_t)src;
    __attribute__((address_space(3))) v4i* l =
        (__attribute__((address_space(3))) v4i*)lds_kv;
    #pragma unroll
    for (int j = 0; j < 8; ++j) {
      const int flat = t + 256 * j;            // 16B chunk id, 0..2047
      const int v = flat >> 4, c = flat & 15;  // row, 16B chunk within row
      __builtin_amdgcn_global_load_async_to_lds_b128(
          g + v * 16 + c, l + v * 17 + c, 0, 0);
    }
#if __has_builtin(__builtin_amdgcn_s_wait_asynccnt)
    __builtin_amdgcn_s_wait_asynccnt(0);
#else
    asm volatile("s_wait_asynccnt 0x0" ::: "memory");
#endif
  }
#else
  for (int i4 = t; i4 < 128 * 32; i4 += 256) {
    const int i = i4 * 4;
    const int v = i >> 7, d = i & 127;
    *(ushort4*)&lds_kv[v * KVSTR + d] = *(const ushort4*)(src + i);
  }
#endif
  if (t < 128) lds_ks[t] = ksum[nh * 128 + t];
  __syncthreads();

  const int lrow = Lbase + wave * 16 + mr;
  const float* qrow = Qg + ((size_t)(n * L_S + lrow) * H_H + h) * D_D;

  f32x8 acc[8];
  #pragma unroll
  for (int i = 0; i < 8; ++i) acc[i] = f32x8{};
  float sp_score = 0.f;   // this lane's half of the row score

  #pragma unroll
  for (int kc = 0; kc < 4; ++kc) {
    const int dbase = kc * 32;
    float4 q0 = *(const float4*)(qrow + dbase + half * 8);
    float4 q1 = *(const float4*)(qrow + dbase + half * 8 + 4);
    float4 q2 = *(const float4*)(qrow + dbase + 16 + half * 8);
    float4 q3 = *(const float4*)(qrow + dbase + 16 + half * 8 + 4);
    const float qf[16] = {q0.x,q0.y,q0.z,q0.w, q1.x,q1.y,q1.z,q1.w,
                          q2.x,q2.y,q2.z,q2.w, q3.x,q3.y,q3.z,q3.w};
    unsigned int w[8];
    #pragma unroll
    for (int p = 0; p < 8; ++p) {   // element pair e = 2p, 2p+1
      float f0 = fmap(qf[2 * p]);
      float f1 = fmap(qf[2 * p + 1]);
      const int k0 = (p < 4) ? (half * 8 + 2 * p) : (16 + half * 8 + 2 * (p - 4));
      sp_score += f0 * lds_ks[dbase + k0] + f1 * lds_ks[dbase + k0 + 1];
      w[p] = pack_bf16(f0, f1);
    }
    Frag a;
    a.i2[0] = make_int4(w[0], w[1], w[2], w[3]);
    a.i2[1] = make_int4(w[4], w[5], w[6], w[7]);
    #pragma unroll
    for (int nt = 0; nt < 8; ++nt) {
      Frag b;
      const unsigned short* br = &lds_kv[(nt * 16 + mr) * KVSTR + dbase + half * 8];
      b.i2[0] = *(const int4*)(br);
      b.i2[1] = *(const int4*)(br + 16);
      acc[nt] = __builtin_amdgcn_wmma_f32_16x16x32_bf16(
          false, a.v, false, b.v, (short)0, acc[nt], false, false);
    }
  }

  // full row score: lanes m and m+16 each hold half the k-range (wave32)
  float score = sp_score + __shfl_xor(sp_score, 16, 32);
  const float th = thr[0];
  #pragma unroll
  for (int r = 0; r < 8; ++r) {
    float sm  = __shfl(score, half * 8 + r, 32);   // row half*8+r of this strip
    float spv = sm > th ? sm : 0.f;
    float z   = 1.0f / (spv + 1e-6f);
    const int M = Lbase + wave * 16 + half * 8 + r;
    float* orow = Og + ((size_t)(n * L_S + M) * H_H + h) * D_D;
    #pragma unroll
    for (int nt = 0; nt < 8; ++nt)
      orow[nt * 16 + mr] = acc[nt][r] * z;
  }
}

// ---------------------------------------------------------------------------
extern "C" void kernel_launch(void* const* d_in, const int* in_sizes, int n_in,
                              void* d_out, int out_size, void* d_ws, size_t ws_size,
                              hipStream_t stream) {
  (void)in_sizes; (void)n_in; (void)out_size; (void)ws_size;
  const float* Q   = (const float*)d_in[0];
  const float* K   = (const float*)d_in[1];
  const float* V   = (const float*)d_in[2];
  const float* thr = (const float*)d_in[3];
  float* O = (float*)d_out;

  // workspace layout (~18 MB): kv partials | ksum partials | KV^T bf16 | Ksum
  float* kv_part   = (float*)d_ws;
  float* ksum_part = kv_part + (size_t)SPLIT * NH * 128 * 128;
  unsigned short* kvT = (unsigned short*)(ksum_part + (size_t)SPLIT * NH * 128);
  float* ksum = (float*)(kvT + (size_t)NH * 128 * 128);

  kv_partial_kernel<<<dim3(SPLIT, NH), 256, 0, stream>>>(K, V, kv_part, ksum_part);
  reduce_kernel<<<dim3(NH), 256, 0, stream>>>(kv_part, ksum_part, kvT, ksum);
  attn_out_kernel<<<dim3(L_S / 128, NH), 256, 0, stream>>>(Q, kvT, ksum, thr, O);
}
